// CNF_13546326852134
// MI455X (gfx1250) — matrix operand order repaired and checked by reading, not compiled, not run on hardware
//
#include <hip/hip_runtime.h>
#include <hip/hip_bf16.h>

typedef __attribute__((ext_vector_type(16))) _Float16 v16h;
typedef __attribute__((ext_vector_type(8)))  _Float16 v8h;
typedef __attribute__((ext_vector_type(8)))  float    v8f;

#define B_SZ        1024
#define N_OBJ       32
#define NPAIR       496
#define ROWS_PER_BLK 128
#define FDIM        128
#define INP         160   /* IN_DIM=146 padded to multiple of 32 */
#define IN_REAL     146
#define PEN_CLIP_F  0.003f

/* ---- LDS layout (bytes) ---- */
#define LDS_XS    0        /* 128x160 f16 = 40960 ; reused as ActB (128x128) */
#define LDS_ACTA  40960    /* 128x128 f16 = 32768 */
#define LDS_W0T   73728    /* 128x160 f16 = 40960 (transposed, padded) */
#define LDS_W1T   114688   /* 128x128 f16 = 32768 (transposed) */
#define LDS_W2T   147456   /* 128x128 f16 = 32768 (transposed) */
#define LDS_W3    180224   /* 128 f16 */
#define LDS_B0    180480   /* 128 f32 */
#define LDS_B1    180992
#define LDS_B2    181504
#define LDS_RES   182016   /* 128 f32 */
#define LDS_TOTAL 182528

/* A fragment: 16x32 f16 tile, row-major LDS, per CDNA5 16-bit A layout.
   lanes 0-15: M=lane, K = k0+0..7 (v0-3) and k0+16..23 (v4-7)
   lanes 16-31: M=lane-16, K = k0+8..15 and k0+24..31            */
__device__ __forceinline__ v16h ldsA(const _Float16* A, int lda, int m0, int k0, int lane) {
    const _Float16* p = A + (m0 + (lane & 15)) * lda + k0 + 8 * (lane >> 4);
    union { v16h v; v8h h[2]; } u;
    u.h[0] = *(const v8h*)(p);
    u.h[1] = *(const v8h*)(p + 16);
    return u.v;
}

/* B fragment: 32x16 f16 tile from TRANSPOSED weights Wt[n][k].
   lanes 0-15: N=lane, K = k0+0..7 and k0+16..23 ; lanes 16-31: N=lane-16, K = k0+8..15 and k0+24..31 */
__device__ __forceinline__ v16h ldsB(const _Float16* Wt, int ldk, int n0, int k0, int lane) {
    const _Float16* p = Wt + (n0 + (lane & 15)) * ldk + k0 + 8 * (lane >> 4);
    union { v16h v; v8h h[2]; } u;
    u.h[0] = *(const v8h*)(p);
    u.h[1] = *(const v8h*)(p + 16);
    return u.v;
}

/* One 128xK @ KxFDIM layer: wave owns N-stripe n0=wave*16, loops 8 M tiles.
   K outer so the B fragment is reused across all 8 WMMAs. All 8 A fragments
   are loaded into registers FIRST (16 back-to-back ds_load_b128), then the
   8 independent WMMAs issue as an uninterrupted chain — removes the
   per-WMMA s_wait_dscnt stalls seen in the previous build. */
__device__ __forceinline__ void mlp_layer(const _Float16* A, int lda, int K,
                                          const _Float16* Wt, int ldk,
                                          const float* bias,
                                          _Float16* D, int ldd,
                                          int wave, int lane) {
    const int n0 = wave * 16;
    v8f acc[8];
#pragma unroll
    for (int m = 0; m < 8; ++m) acc[m] = (v8f){};
#pragma unroll
    for (int k0 = 0; k0 < K; k0 += 32) {
        v16h bf = ldsB(Wt, ldk, n0, k0, lane);
        v16h af[8];
#pragma unroll
        for (int m = 0; m < 8; ++m)
            af[m] = ldsA(A, lda, m * 16, k0, lane);
#pragma unroll
        for (int m = 0; m < 8; ++m)
            acc[m] = __builtin_amdgcn_wmma_f32_16x16x32_f16(
                false, af[m], false, bf, (short)0, acc[m], false, false);
    }
    const int nCol = n0 + (lane & 15);
    const int mOff = (lane >> 4) * 8;
    const float bv = bias[nCol];
#pragma unroll
    for (int m = 0; m < 8; ++m) {
#pragma unroll
        for (int r = 0; r < 8; ++r) {
            float v = acc[m][r] + bv;
            v = v > 0.f ? v : 0.f;
            D[(m * 16 + mOff + r) * ldd + nCol] = (_Float16)v;
        }
    }
}

__global__ void __launch_bounds__(256)
CNF_pen_mlp_wmma(const float* __restrict__ objt, const float* __restrict__ pos,
                 const float* __restrict__ quat,
                 const float* __restrict__ W0, const float* __restrict__ b0,
                 const float* __restrict__ W1, const float* __restrict__ b1,
                 const float* __restrict__ W2, const float* __restrict__ b2,
                 const float* __restrict__ W3, const float* __restrict__ b3,
                 float* __restrict__ out)
{
    extern __shared__ char smem[];
    _Float16* Xs   = (_Float16*)(smem + LDS_XS);
    _Float16* ActA = (_Float16*)(smem + LDS_ACTA);
    _Float16* W0t  = (_Float16*)(smem + LDS_W0T);
    _Float16* W1t  = (_Float16*)(smem + LDS_W1T);
    _Float16* W2t  = (_Float16*)(smem + LDS_W2T);
    _Float16* w3s  = (_Float16*)(smem + LDS_W3);
    float*    b0s  = (float*)(smem + LDS_B0);
    float*    b1s  = (float*)(smem + LDS_B1);
    float*    b2s  = (float*)(smem + LDS_B2);
    float*    resB = (float*)(smem + LDS_RES);

    const int tid = threadIdx.x;

    /* ---- weights -> LDS, transposed, f32 -> f16, W0 zero-padded K 146..159 ---- */
    for (int idx = tid; idx < FDIM * INP; idx += 256) {
        int n = idx / INP, k = idx - n * INP;
        W0t[idx] = (_Float16)((k < IN_REAL) ? W0[k * FDIM + n] : 0.f);
    }
    for (int idx = tid; idx < FDIM * FDIM; idx += 256) {
        int n = idx >> 7, k = idx & 127;
        W1t[idx] = (_Float16)W1[k * FDIM + n];
        W2t[idx] = (_Float16)W2[k * FDIM + n];
    }
    if (tid < FDIM) {
        w3s[tid] = (_Float16)W3[tid];
        b0s[tid] = b0[tid];
        b1s[tid] = b1[tid];
        b2s[tid] = b2[tid];
    }

    /* ---- build 128 feature rows (2 threads/row: freq halves) ---- */
    {
        const int row  = tid & 127;
        const int half = tid >> 7;
        const int gRow = blockIdx.x * ROWS_PER_BLK + row;
        const int b    = gRow / (NPAIR * 2);
        const int rem  = gRow - b * (NPAIR * 2);
        const int p    = rem >> 1;
        const int dir  = rem & 1;
        int i = 0, pp = p;
        while (pp >= (N_OBJ - 1 - i)) { pp -= (N_OBJ - 1 - i); ++i; }
        const int j = i + 1 + pp;
        const int a = dir ? j : i;
        const int c = dir ? i : j;

        const float* pa = pos  + (b * N_OBJ + a) * 3;
        const float* pc = pos  + (b * N_OBJ + c) * 3;
        const float* qa = quat + (b * N_OBJ + a) * 4;
        const float* qc = quat + (b * N_OBJ + c) * 4;

        float pdv[3] = { pc[0] - pa[0], pc[1] - pa[1], pc[2] - pa[2] };
        float vax = qa[0], vay = qa[1], vaz = qa[2], wa = qa[3];
        float vcx = qc[0], vcy = qc[1], vcz = qc[2], wc = qc[3];
        /* qinv(qa) * qc : v = wa*vc - wc*va - cross(va,vc) ; w = wa*wc + dot(va,vc) */
        float lqv[4];
        lqv[0] = wa * vcx - wc * vax - (vay * vcz - vaz * vcy);
        lqv[1] = wa * vcy - wc * vay - (vaz * vcx - vax * vcz);
        lqv[2] = wa * vcz - wc * vaz - (vax * vcy - vay * vcx);
        lqv[3] = wa * wc + vax * vcx + vay * vcy + vaz * vcz;

        _Float16* xr = Xs + row * INP;
        if (half == 0) {
            const float* ta = objt + (b * N_OBJ + a) * 3;
            const float* tc = objt + (b * N_OBJ + c) * 3;
            xr[0] = (_Float16)ta[0]; xr[1] = (_Float16)ta[1]; xr[2] = (_Float16)ta[2];
            xr[3] = (_Float16)tc[0]; xr[4] = (_Float16)tc[1]; xr[5] = (_Float16)tc[2];
#pragma unroll
            for (int k = IN_REAL; k < INP; ++k) xr[k] = (_Float16)0.f;
        }
        /* pe layout: pd -> feat[6 + e*6 + {0:cos,1:sin}*3 + d]; lq -> feat[66 + e*8 + c*4 + d] */
        float f = 3.14159265358979323846f * (float)(1 << (half * 5));
#pragma unroll
        for (int e = 0; e < 5; ++e) {
            const int eg = half * 5 + e;
#pragma unroll
            for (int d = 0; d < 3; ++d) {
                float s, co;
                __sincosf(pdv[d] * f, &s, &co);
                xr[6 + eg * 6 + d]     = (_Float16)co;
                xr[6 + eg * 6 + 3 + d] = (_Float16)s;
            }
#pragma unroll
            for (int d = 0; d < 4; ++d) {
                float s, co;
                __sincosf(lqv[d] * f, &s, &co);
                xr[66 + eg * 8 + d]     = (_Float16)co;
                xr[66 + eg * 8 + 4 + d] = (_Float16)s;
            }
            f *= 2.f;
        }
    }
    __syncthreads();

    const int wave = tid >> 5;
    const int lane = tid & 31;

    /* Layer 0: Xs(128x160) @ W0t -> ActA */
    mlp_layer(Xs, INP, INP, W0t, INP, b0s, ActA, FDIM, wave, lane);
    __syncthreads();
    /* Layer 1: ActA -> ActB (reuses Xs region) */
    _Float16* ActB = Xs;
    mlp_layer(ActA, FDIM, FDIM, W1t, FDIM, b1s, ActB, FDIM, wave, lane);
    __syncthreads();
    /* Layer 2: ActB -> ActA */
    mlp_layer(ActB, FDIM, FDIM, W2t, FDIM, b2s, ActA, FDIM, wave, lane);
    __syncthreads();

    /* Layer 3: 128 -> 1, tanh clip */
    if (tid < FDIM) {
        float acc = b3[0];
#pragma unroll 8
        for (int k = 0; k < FDIM; ++k)
            acc += (float)ActA[tid * FDIM + k] * (float)w3s[k];
        resB[tid] = PEN_CLIP_F * tanhf(acc);
    }
    __syncthreads();

    /* mean over the 2 direction rows (consecutive) */
    if (tid < 64) {
        out[blockIdx.x * 64 + tid] = 0.5f * (resB[2 * tid] + resB[2 * tid + 1]);
    }
}

extern "C" void kernel_launch(void* const* d_in, const int* in_sizes, int n_in,
                              void* d_out, int out_size, void* d_ws, size_t ws_size,
                              hipStream_t stream) {
    (void)in_sizes; (void)n_in; (void)out_size; (void)d_ws; (void)ws_size;
    const float* objt = (const float*)d_in[0];
    /* d_in[1] = gparam: unused by the reference */
    const float* pos  = (const float*)d_in[2];
    const float* quat = (const float*)d_in[3];
    const float* W0   = (const float*)d_in[4];
    const float* b0   = (const float*)d_in[5];
    const float* W1   = (const float*)d_in[6];
    const float* b1   = (const float*)d_in[7];
    const float* W2   = (const float*)d_in[8];
    const float* b2   = (const float*)d_in[9];
    const float* W3   = (const float*)d_in[10];
    const float* b3   = (const float*)d_in[11];
    float* out = (float*)d_out;

    const int totalRows = B_SZ * NPAIR * 2;             /* 1,015,808 */
    const int nBlocks   = totalRows / ROWS_PER_BLK;     /* 7936 exactly */

    static_assert(LDS_TOTAL <= 320 * 1024, "fits in WGP LDS");
    (void)hipFuncSetAttribute((const void*)CNF_pen_mlp_wmma,
                              hipFuncAttributeMaxDynamicSharedMemorySize, LDS_TOTAL);

    CNF_pen_mlp_wmma<<<nBlocks, 256, LDS_TOTAL, stream>>>(
        objt, pos, quat, W0, b0, W1, b1, W2, b2, W3, b3, out);
}
